// Tree_Net_74663711473669
// MI455X (gfx1250) — matrix operand order repaired and checked by reading, not compiled, not run on hardware
//
#include <hip/hip_runtime.h>
#include <hip/hip_bf16.h>
#include <math.h>

#define B_   256
#define L_   64
#define D_   1024
#define V_   50000
#define C_   500
#define N_   127   // 2L-1
#define S_   63    // L-1
#define CPAD 512   // C padded to multiple of 128 for the GEMM tile

typedef __attribute__((ext_vector_type(16))) __bf16 v16bf;
typedef __attribute__((ext_vector_type(8)))  float  v8f;

// ---------------------------------------------------------------------------
// K0: convert lin_w (C x D fp32) -> bf16, padded with zero rows up to CPAD
// ---------------------------------------------------------------------------
__global__ void convert_w_kernel(const float* __restrict__ w, __bf16* __restrict__ wb) {
    int idx = blockIdx.x * 256 + threadIdx.x;   // over CPAD*D
    int c = idx >> 10;
    int k = idx & (D_ - 1);
    float v = (c < C_) ? w[c * D_ + k] : 0.0f;
    wb[idx] = (__bf16)v;
}

// ---------------------------------------------------------------------------
// K1: gather leaf embeddings, mask, L2-normalize; zero internal-node rows.
// One 256-thread block per (b, n) row; 4 floats per thread.
// ---------------------------------------------------------------------------
__global__ void leaf_kernel(const int* __restrict__ ids, const int* __restrict__ mask,
                            const float* __restrict__ emb, float* __restrict__ vec) {
    int row = blockIdx.x;            // 0 .. B*N-1
    int b = row / N_;
    int n = row - b * N_;
    float* dst = vec + (size_t)row * D_;
    int t = threadIdx.x;             // 256 threads
    __shared__ float red[256];

    if (n >= L_) {                   // internal node: zero-init (ws is poisoned)
        float4 z = {0.f, 0.f, 0.f, 0.f};
        ((float4*)dst)[t] = z;
        return;
    }
    int id = ids[b * L_ + n];
    int m  = mask[b * L_ + n];
    const float* src = emb + (size_t)id * D_;
    float4 v = ((const float4*)src)[t];
    if (m <= 0) { v.x = 0.f; v.y = 0.f; v.z = 0.f; v.w = 0.f; }
    red[t] = v.x * v.x + v.y * v.y + v.z * v.z + v.w * v.w;
    __syncthreads();
    for (int off = 128; off > 0; off >>= 1) {
        if (t < off) red[t] += red[t + off];
        __syncthreads();
    }
    float inv = 1.0f / (sqrtf(red[0]) + 1e-6f);
    float4 o = {v.x * inv, v.y * inv, v.z * inv, v.w * inv};
    ((float4*)dst)[t] = o;
}

// ---------------------------------------------------------------------------
// K2: sequential tree scan. One 256-thread block per batch element.
// c[k] = sum_j a[j] * b[(j+k) % D]  (circular cross-correlation), then
// normalize by (||c|| + 1e-6). b is duplicated in LDS to avoid the modulo.
// Each thread owns k, k+256, k+512, k+768 (4 FMA chains per broadcast a[j]).
// ---------------------------------------------------------------------------
__global__ void scan_kernel(const int* __restrict__ comp, float* __restrict__ vec) {
    int b = blockIdx.x;
    int t = threadIdx.x;             // 256 threads
    __shared__ float a_sh[D_];
    __shared__ float b_sh[2 * D_];
    __shared__ float red[256];
    float* base = vec + (size_t)b * N_ * D_;

    for (int s = 0; s < S_; ++s) {
        const int* info = comp + ((size_t)b * S_ + s) * 3;
        int il = info[0], ir = info[1], io = info[2];
        const float* lv = base + (size_t)il * D_;
        const float* rv = base + (size_t)ir * D_;
#pragma unroll
        for (int i = 0; i < 4; ++i) {
            int idx = t + i * 256;
            a_sh[idx] = lv[idx];
            float r = rv[idx];
            b_sh[idx] = r;
            b_sh[idx + D_] = r;
        }
        __syncthreads();

        float acc0 = 0.f, acc1 = 0.f, acc2 = 0.f, acc3 = 0.f;
#pragma unroll 8
        for (int j = 0; j < D_; ++j) {
            float av = a_sh[j];
            acc0 = fmaf(av, b_sh[j + t      ], acc0);
            acc1 = fmaf(av, b_sh[j + t + 256], acc1);
            acc2 = fmaf(av, b_sh[j + t + 512], acc2);
            acc3 = fmaf(av, b_sh[j + t + 768], acc3);
        }

        red[t] = acc0 * acc0 + acc1 * acc1 + acc2 * acc2 + acc3 * acc3;
        __syncthreads();
        for (int off = 128; off > 0; off >>= 1) {
            if (t < off) red[t] += red[t + off];
            __syncthreads();
        }
        float inv = 1.0f / (sqrtf(red[0]) + 1e-6f);

        float* out = base + (size_t)io * D_;
        out[t]       = acc0 * inv;
        out[t + 256] = acc1 * inv;
        out[t + 512] = acc2 * inv;
        out[t + 768] = acc3 * inv;
        __threadfence();             // make our global writes visible block-wide
        __syncthreads();
    }
}

// ---------------------------------------------------------------------------
// K3: logits GEMM via v_wmma_f32_16x16x32_bf16 + bias + sigmoid.
// A = vec (M=32512 x K=1024 fp32, converted to bf16 in-register)
// B = wb  (CPAD x K row-major bf16; class rows are the WMMA B columns)
// Wave tile: 16(M) x 64(N) via 4 accumulators. Block: 8 waves = 64(M) x 128(N).
// Fragment k order is contiguous (k = 16*hi + s), the same for A and B.
// ---------------------------------------------------------------------------
__global__ void gemm_kernel(const float* __restrict__ A, const __bf16* __restrict__ Wb,
                            const float* __restrict__ bias, float* __restrict__ out) {
    int wave = threadIdx.x >> 5;
    int lane = threadIdx.x & 31;
    int lh   = lane & 15;
    int hi   = lane >> 4;                    // 0: lanes 0-15, 1: lanes 16-31

    int m0 = blockIdx.x * 64 + (wave & 3) * 16;
    int n0 = blockIdx.y * 128 + (wave >> 2) * 64;

    const float* Ap = A + (size_t)(m0 + lh) * D_;
    int aoff = hi ? 16 : 0;                  // A: K[aoff..aoff+15]

    v8f cacc[4];
#pragma unroll
    for (int f = 0; f < 4; ++f) cacc[f] = (v8f){0.f,0.f,0.f,0.f,0.f,0.f,0.f,0.f};

    for (int kb = 0; kb < D_; kb += 32) {
        float4 x0 = *(const float4*)(Ap + kb + aoff);
        float4 x1 = *(const float4*)(Ap + kb + aoff + 4);
        float4 x2 = *(const float4*)(Ap + kb + aoff + 8);
        float4 x3 = *(const float4*)(Ap + kb + aoff + 12);
        v16bf a;
        a[0]  = (__bf16)x0.x; a[1]  = (__bf16)x0.y; a[2]  = (__bf16)x0.z; a[3]  = (__bf16)x0.w;
        a[4]  = (__bf16)x1.x; a[5]  = (__bf16)x1.y; a[6]  = (__bf16)x1.z; a[7]  = (__bf16)x1.w;
        a[8]  = (__bf16)x2.x; a[9]  = (__bf16)x2.y; a[10] = (__bf16)x2.z; a[11] = (__bf16)x2.w;
        a[12] = (__bf16)x3.x; a[13] = (__bf16)x3.y; a[14] = (__bf16)x3.z; a[15] = (__bf16)x3.w;

        // B fragment f: columns n0+16f .. n0+16f+15 ; lane lh -> class row,
        // elements 0..15 = K[kb + hi*16 .. +15]
        const __bf16* bp = Wb + (size_t)(n0 + lh) * D_ + kb + hi * 16;
        v16bf b0 = *(const v16bf*)(bp);
        v16bf b1 = *(const v16bf*)(bp + (size_t)16 * D_);
        v16bf b2 = *(const v16bf*)(bp + (size_t)32 * D_);
        v16bf b3 = *(const v16bf*)(bp + (size_t)48 * D_);

        cacc[0] = __builtin_amdgcn_wmma_f32_16x16x32_bf16(false, a, false, b0,
                                                          (short)0, cacc[0], false, false);
        cacc[1] = __builtin_amdgcn_wmma_f32_16x16x32_bf16(false, a, false, b1,
                                                          (short)0, cacc[1], false, false);
        cacc[2] = __builtin_amdgcn_wmma_f32_16x16x32_bf16(false, a, false, b2,
                                                          (short)0, cacc[2], false, false);
        cacc[3] = __builtin_amdgcn_wmma_f32_16x16x32_bf16(false, a, false, b3,
                                                          (short)0, cacc[3], false, false);
    }

    // Epilogue: D element (m,n): VGPR r -> m = m0 + r + hi*8, n = n0 + 16f + lh
#pragma unroll
    for (int f = 0; f < 4; ++f) {
        int n = n0 + f * 16 + lh;
        if (n >= C_) continue;
        float bv = bias[n];
#pragma unroll
        for (int r = 0; r < 8; ++r) {
            int m = m0 + r + hi * 8;
            float v = cacc[f][r] + bv;
            out[(size_t)m * C_ + n] = 1.0f / (1.0f + expf(-v));
        }
    }
}

// ---------------------------------------------------------------------------
extern "C" void kernel_launch(void* const* d_in, const int* in_sizes, int n_in,
                              void* d_out, int out_size, void* d_ws, size_t ws_size,
                              hipStream_t stream) {
    const int*   ids  = (const int*)d_in[0];     // (B, L)
    const int*   mask = (const int*)d_in[1];     // (B, L)
    const int*   comp = (const int*)d_in[2];     // (B, S, 3)
    const float* emb  = (const float*)d_in[3];   // (V, D)
    const float* lw   = (const float*)d_in[4];   // (C, D)
    const float* lb   = (const float*)d_in[5];   // (C,)
    float* out = (float*)d_out;                  // (B, N, C)

    char* ws = (char*)d_ws;
    float*  vec = (float*)ws;                                     // B*N*D fp32
    __bf16* wb  = (__bf16*)(ws + (size_t)B_ * N_ * D_ * sizeof(float)); // CPAD*D bf16

    convert_w_kernel<<<(CPAD * D_) / 256, 256, 0, stream>>>(lw, wb);
    leaf_kernel<<<B_ * N_, 256, 0, stream>>>(ids, mask, emb, vec);
    scan_kernel<<<B_, 256, 0, stream>>>(comp, vec);
    gemm_kernel<<<dim3((B_ * N_) / 64, CPAD / 128), 256, 0, stream>>>(vec, wb, lb, out);
}
